// GRNNTransformGated_5463198401279
// MI455X (gfx1250) — matrix operand order repaired and checked by reading, not compiled, not run on hardware
//
#include <hip/hip_runtime.h>
#include <hip/hip_bf16.h>

typedef _Float16 h16;
typedef __attribute__((ext_vector_type(16))) _Float16 v16h;
typedef __attribute__((ext_vector_type(8)))  _Float16 h8;
typedef __attribute__((ext_vector_type(8)))  float    v8f;

#define LVLS 13
#define H 64
#define F 7

// ---------------------------------------------------------------------------
// One-shot weight conversion fp32 -> f16 (weights then live in L2 for the
// whole tree sweep; 0.23 MB << 192 MB L2).
// ---------------------------------------------------------------------------
__global__ void cvt_w_kernel(const float* __restrict__ Wr,
                             const float* __restrict__ Wh,
                             const float* __restrict__ Wz,
                             h16* __restrict__ wr, h16* __restrict__ wh,
                             h16* __restrict__ wz) {
    int i = blockIdx.x * 256 + threadIdx.x;
    if (i < 192 * 192) wr[i] = (h16)Wr[i];
    if (i < 64 * 192)  wh[i] = (h16)Wh[i];
    if (i < 256 * 256) wz[i] = (h16)Wz[i];
}

// ---------------------------------------------------------------------------
// Leaf level (j = 12): u = relu(c @ W_u^T + b_u). K=7 -> VALU, store f16.
// ---------------------------------------------------------------------------
__global__ void leaf_u_kernel(const float* __restrict__ contents,
                              const float* __restrict__ W_u,
                              const float* __restrict__ b_u,
                              h16* __restrict__ up, long long off,
                              long long nrows) {
    long long e = (long long)blockIdx.x * 256 + threadIdx.x;
    if (e >= nrows * H) return;
    long long row = e >> 6;
    int col = (int)(e & 63);
    const float* c = contents + (off + row) * F;
    float s = b_u[col];
#pragma unroll
    for (int k = 0; k < F; ++k) s += c[k] * W_u[col * F + k];
    up[(off + row) * H + col] = (h16)fmaxf(s, 0.f);
}

// ---------------------------------------------------------------------------
// Interior level kernel: one block handles 64 rows of level j.
//   LDS: hhu (64x192 f16), rh (64x192 f16), hH (64x64 f16), z (64x256 f32)
//   = 120 KB (CDNA5 WGP has 320 KB).
// All matmuls via v_wmma_f32_16x16x32_f16; tile loops are wave-uniform so
// EXEC is all-ones at every WMMA (ISA requirement).
// ---------------------------------------------------------------------------
__global__ __launch_bounds__(256) void level_kernel(
    const float* __restrict__ contents,
    const float* __restrict__ W_u, const float* __restrict__ b_u,
    const h16* __restrict__ Wr, const float* __restrict__ b_r,
    const h16* __restrict__ Wh, const float* __restrict__ b_h,
    const h16* __restrict__ Wz, const float* __restrict__ b_z,
    h16* __restrict__ up, long long lvlOff, long long childOff,
    float* __restrict__ outp)
{
    __shared__ h16  sh_hhu[64 * 192];   // [h_L | h_R | u]
    __shared__ h16  sh_rh [64 * 192];   // sigmoid(r) * hhu
    __shared__ h16  sh_hH [64 * 64];
    __shared__ float sh_z [64 * 256];

    const int tid  = threadIdx.x;
    const int lane = tid & 31;
    const int wave = tid >> 5;
    const long long rowBase = (long long)blockIdx.x * 64;

    // ---- (a) u = relu(c @ W_u^T + b_u) into hhu[:, 128:192] -------------
    for (int e = tid; e < 64 * 64; e += 256) {
        int m = e >> 6, col = e & 63;
        const float* c = contents + (lvlOff + rowBase + m) * F;
        float s = b_u[col];
#pragma unroll
        for (int k = 0; k < F; ++k) s += c[k] * W_u[col * F + k];
        sh_hhu[m * 192 + 128 + col] = (h16)fmaxf(s, 0.f);
    }
    // ---- (b) h_L = up[0::2], h_R = up[1::2] into hhu[:, 0:128] ----------
    for (int e = tid; e < 64 * 16; e += 256) {
        int m = e >> 4;
        int c0 = (e & 15) * 8;                 // column chunk 0..120
        long long child = childOff + 2 * (rowBase + m) + (c0 >= 64 ? 1 : 0);
        int cc = c0 & 63;
        h8 v = *(const h8*)(up + child * H + cc);
        *(h8*)(&sh_hhu[m * 192 + c0]) = v;
    }
    __syncthreads();

    // A-fragment from LDS, 16-bit A 16x32 layout:
    //   lanes 0-15: K {k0..k0+7, k0+16..k0+23}; lanes 16-31: +8.
    auto loadA = [&](const h16* base, int stride, int m0, int k0) -> v16h {
        int m   = m0 + (lane & 15);
        int klo = k0 + ((lane < 16) ? 0 : 8);
        h8 lo = *(const h8*)(base + m * stride + klo);
        h8 hi = *(const h8*)(base + m * stride + klo + 16);
        v16h a;
#pragma unroll
        for (int i = 0; i < 8; ++i) { a[i] = lo[i]; a[8 + i] = hi[i]; }
        return a;
    };
    // B-fragment from row-major f16 weights W[N][K]: B[k][n] = W[n][k];
    // lane holds K {k0..k0+7, k0+16..k0+23} (+8 for lanes 16-31) for column n.
    auto loadB = [&](const h16* W, int K, int n0, int k0) -> v16h {
        int n  = n0 + (lane & 15);
        int kb = k0 + ((lane < 16) ? 0 : 8);
        h8 lo = *(const h8*)(W + n * K + kb);
        h8 hi = *(const h8*)(W + n * K + kb + 16);
        v16h b;
#pragma unroll
        for (int i = 0; i < 8; ++i) { b[i] = lo[i]; b[8 + i] = hi[i]; }
        return b;
    };

    // ---- (c) r = sigmoid(hhu @ W_r^T + b_r); rh = r * hhu ---------------
    // 4 row-tiles x 12 col-tiles = 48 tiles, K = 192 (6 steps of 32).
    for (int t = wave; t < 48; t += 8) {
        int m0 = (t / 12) * 16, n0 = (t % 12) * 16;
        v8f acc = {};
#pragma unroll
        for (int kk = 0; kk < 6; ++kk) {
            v16h a = loadA(sh_hhu, 192, m0, kk * 32);
            v16h b = loadB(Wr, 192, n0, kk * 32);
            acc = __builtin_amdgcn_wmma_f32_16x16x32_f16(
                false, a, false, b, (short)0, acc, false, false);
        }
        int n = n0 + (lane & 15);
        float bias = b_r[n];
        int mbase = m0 + ((lane & 16) ? 8 : 0);
#pragma unroll
        for (int p = 0; p < 8; ++p) {
            int m = mbase + p;
            float x  = acc[p] + bias;
            float sg = 1.f / (1.f + __expf(-x));
            float hv = (float)sh_hhu[m * 192 + n];
            sh_rh[m * 192 + n] = (h16)(sg * hv);
        }
    }
    __syncthreads();

    // ---- (d) h_H = relu(rh @ W_h^T + b_h): 4x4 tiles, K = 192 -----------
    for (int t = wave; t < 16; t += 8) {
        int m0 = (t / 4) * 16, n0 = (t % 4) * 16;
        v8f acc = {};
#pragma unroll
        for (int kk = 0; kk < 6; ++kk) {
            v16h a = loadA(sh_rh, 192, m0, kk * 32);
            v16h b = loadB(Wh, 192, n0, kk * 32);
            acc = __builtin_amdgcn_wmma_f32_16x16x32_f16(
                false, a, false, b, (short)0, acc, false, false);
        }
        int n = n0 + (lane & 15);
        float bias = b_h[n];
        int mbase = m0 + ((lane & 16) ? 8 : 0);
#pragma unroll
        for (int p = 0; p < 8; ++p)
            sh_hH[(mbase + p) * 64 + n] = (h16)fmaxf(acc[p] + bias, 0.f);
    }
    __syncthreads();

    // ---- (e) z = [h_H | hhu] @ W_z^T + b_z: 4x16 tiles, K = 256 ---------
    // K split: k<64 reads A from hH buffer, k>=64 from hhu (avoids a copy).
    for (int t = wave; t < 64; t += 8) {
        int m0 = (t / 16) * 16, n0 = (t % 16) * 16;
        v8f acc = {};
#pragma unroll
        for (int kk = 0; kk < 8; ++kk) {
            int k0 = kk * 32;
            v16h a = (k0 < 64) ? loadA(sh_hH, 64, m0, k0)
                               : loadA(sh_hhu, 192, m0, k0 - 64);
            v16h b = loadB(Wz, 256, n0, k0);
            acc = __builtin_amdgcn_wmma_f32_16x16x32_f16(
                false, a, false, b, (short)0, acc, false, false);
        }
        int n = n0 + (lane & 15);
        float bias = b_z[n];
        int mbase = m0 + ((lane & 16) ? 8 : 0);
#pragma unroll
        for (int p = 0; p < 8; ++p)
            sh_z[(mbase + p) * 256 + n] = acc[p] + bias;
    }
    __syncthreads();

    // ---- (f) softmax over 4 gates + combine; emit up (f16) / out (f32) --
    for (int e = tid; e < 64 * 64; e += 256) {
        int m = e >> 6, c = e & 63;
        float z0 = sh_z[m * 256 + c];
        float z1 = sh_z[m * 256 + 64 + c];
        float z2 = sh_z[m * 256 + 128 + c];
        float z3 = sh_z[m * 256 + 192 + c];
        float mx = fmaxf(fmaxf(z0, z1), fmaxf(z2, z3));
        float e0 = __expf(z0 - mx), e1 = __expf(z1 - mx);
        float e2 = __expf(z2 - mx), e3 = __expf(z3 - mx);
        float inv = 1.f / (e0 + e1 + e2 + e3);
        float hH = (float)sh_hH[m * 64 + c];
        float hL = (float)sh_hhu[m * 192 + c];
        float hR = (float)sh_hhu[m * 192 + 64 + c];
        float uu = (float)sh_hhu[m * 192 + 128 + c];
        float res = (e0 * hH + e1 * hL + e2 * hR + e3 * uu) * inv;
        up[(lvlOff + rowBase + m) * H + c] = (h16)res;
        if (outp) outp[(rowBase + m) * H + c] = res;
    }
}

// ---------------------------------------------------------------------------
extern "C" void kernel_launch(void* const* d_in, const int* in_sizes, int n_in,
                              void* d_out, int out_size, void* d_ws, size_t ws_size,
                              hipStream_t stream) {
    (void)in_sizes; (void)n_in; (void)out_size; (void)ws_size;
    const float* contents = (const float*)d_in[0];
    const float* W_u = (const float*)d_in[1];
    const float* b_u = (const float*)d_in[2];
    const float* W_r = (const float*)d_in[3];
    const float* b_r = (const float*)d_in[4];
    const float* W_h = (const float*)d_in[5];
    const float* b_h = (const float*)d_in[6];
    const float* W_z = (const float*)d_in[7];
    const float* b_z = (const float*)d_in[8];
    float* out = (float*)d_out;

    long long OFFS[LVLS + 1];
    for (int j = 0; j <= LVLS; ++j) OFFS[j] = 64LL * ((1LL << j) - 1);

    // Workspace layout: up-tree (f16, same offsets as contents) + f16 weights.
    char* ws = (char*)d_ws;
    size_t upBytes = ((size_t)OFFS[LVLS] * H * sizeof(h16) + 255) & ~(size_t)255;
    size_t wrBytes = ((size_t)192 * 192 * sizeof(h16) + 255) & ~(size_t)255;
    size_t whBytes = ((size_t)64 * 192 * sizeof(h16) + 255) & ~(size_t)255;
    h16* up = (h16*)ws;
    h16* wr = (h16*)(ws + upBytes);
    h16* wh = (h16*)(ws + upBytes + wrBytes);
    h16* wz = (h16*)(ws + upBytes + wrBytes + whBytes);

    cvt_w_kernel<<<(256 * 256 + 255) / 256, 256, 0, stream>>>(W_r, W_h, W_z, wr, wh, wz);

    {   // leaf level j = 12
        long long n12 = 64LL << 12;
        long long threads = n12 * H;
        leaf_u_kernel<<<(int)((threads + 255) / 256), 256, 0, stream>>>(
            contents, W_u, b_u, up, OFFS[12], n12);
    }
    // interior levels j = 11 .. 0 (strictly sequential, same stream)
    for (int j = LVLS - 2; j >= 0; --j) {
        int nblocks = (int)((64LL << j) / 64);
        level_kernel<<<nblocks, 256, 0, stream>>>(
            contents, W_u, b_u, wr, b_r, wh, b_h, wz, b_z,
            up, OFFS[j], OFFS[j + 1], (j == 0) ? out : nullptr);
    }
}
